// GraphAttention_11244224380973
// MI455X (gfx1250) — matrix operand order repaired and checked by reading, not compiled, not run on hardware
//
#include <hip/hip_runtime.h>
#include <hip/hip_bf16.h>

#define N_NODES 8192
#define F_IN    128
#define F_OUT   64
#define HEADS   2
#define ALPHA   0.2f
#define LOG2E   1.4426950408889634f
#define PARTS   4
#define COLS_PER_PART (N_NODES / PARTS)   // 2048
#define CHUNKS  (COLS_PER_PART / 32)      // 64

typedef __attribute__((ext_vector_type(16))) __bf16 v16bf;
typedef __attribute__((ext_vector_type(8)))  float  v8f;

// ---- workspace layout (bytes) ----
#define OFF_FEATST 0u
#define SZ_FEATST  (HEADS * F_OUT * N_NODES * 2u)       // bf16 feats^T  (2 MB)
#define OFF_AS     (OFF_FEATST + SZ_FEATST)
#define SZ_SCAL    (HEADS * N_NODES * 4u)               // 64 KB each
#define OFF_AN     (OFF_AS + SZ_SCAL)
#define OFF_M      (OFF_AN + SZ_SCAL)
#define OFF_IZ     (OFF_M  + SZ_SCAL)
#define OFF_MASK   (OFF_IZ + SZ_SCAL)
#define SZ_MASK    (N_NODES * (N_NODES / 32) * 4u)      // 8 MB bitmask
#define OFF_ACC    (OFF_MASK + SZ_MASK)
#define SZ_ACC     (N_NODES * F_OUT * 4u)               // 2 MB head-sum accumulator

// ------------------------------------------------------------------
// Kernel 1: feats = X @ W per head; emit feats^T (bf16, K-contiguous),
//           a_s/a_n pre-scaled by log2(e) so softmax can use raw v_exp_f32.
// ------------------------------------------------------------------
__global__ void __launch_bounds__(128) gat_proj(
    const float* __restrict__ X, const float* __restrict__ W,
    const float* __restrict__ a_self, const float* __restrict__ a_neigh,
    __bf16* __restrict__ featsT, float* __restrict__ a_s, float* __restrict__ a_n) {
  __shared__ float s_x[F_IN];
  __shared__ float s_ra[128], s_rb[128];
  const int n = blockIdx.x, t = threadIdx.x;
  s_x[t] = X[(size_t)n * F_IN + t];
  __syncthreads();
  const int h = t >> 6, o = t & 63;
  float acc = 0.f;
#pragma unroll 8
  for (int f = 0; f < F_IN; ++f)
    acc = fmaf(s_x[f], W[(size_t)(h * F_IN + f) * F_OUT + o], acc);
  featsT[(size_t)(h * F_OUT + o) * N_NODES + n] = (__bf16)acc;
  s_ra[t] = acc * a_self[h * F_OUT + o];
  s_rb[t] = acc * a_neigh[h * F_OUT + o];
  __syncthreads();
  for (int s = 32; s >= 1; s >>= 1) {
    if (o < s) { s_ra[t] += s_ra[t + s]; s_rb[t] += s_rb[t + s]; }
    __syncthreads();
  }
  if (o == 0) {
    a_s[h * N_NODES + n] = s_ra[t] * LOG2E;   // pre-scaled: exp(x) == exp2(x')
    a_n[h * N_NODES + n] = s_rb[t] * LOG2E;
  }
}

// ------------------------------------------------------------------
// Kernel 2: dense A (256 MB, read exactly once, non-temporal) -> 8 MB bitmask
// ------------------------------------------------------------------
__global__ void __launch_bounds__(256) gat_mask(
    const float* __restrict__ A, unsigned* __restrict__ mask) {
  const size_t g = (size_t)blockIdx.x * 256u + threadIdx.x;   // element index
  const float a = __builtin_nontemporal_load(&A[g]);
  const unsigned bal = __builtin_amdgcn_ballot_w32(a != 0.0f);
  if ((threadIdx.x & 31u) == 0u) mask[g >> 5] = bal;
}

// ------------------------------------------------------------------
// Kernel 3: exact per-row softmax stats (base-2 domain).
//   m_i = lr(a_s_i + max_{j in N(i)} a_n_j)   (lr monotone, commutes w/ scale)
//   Z_i = sum_{j in N(i)} exp2(lr(a_s_i + a_n_j) - m_i)
// ------------------------------------------------------------------
__global__ void __launch_bounds__(32) gat_rowstats(
    const unsigned* __restrict__ mask, const float* __restrict__ a_s,
    const float* __restrict__ a_n, float* __restrict__ mrow, float* __restrict__ invZ) {
  const int i = blockIdx.x, h = blockIdx.y, lane = threadIdx.x;
  const unsigned* mrp = mask + (size_t)i * (N_NODES / 32);
  const float* an = a_n + (size_t)h * N_NODES;
  float mx = -1e30f;
  for (int w = lane; w < N_NODES / 32; w += 32) {
    unsigned bits = mrp[w];
    while (bits) {
      const int b = __builtin_ctz(bits); bits &= bits - 1u;
      mx = fmaxf(mx, an[w * 32 + b]);
    }
  }
  for (int off = 16; off >= 1; off >>= 1) mx = fmaxf(mx, __shfl_xor(mx, off, 32));
  const float asi = a_s[h * N_NODES + i];
  const float e0 = asi + mx;
  const float m = fmaxf(e0, ALPHA * e0);
  float z = 0.f;
  for (int w = lane; w < N_NODES / 32; w += 32) {
    unsigned bits = mrp[w];
    while (bits) {
      const int b = __builtin_ctz(bits); bits &= bits - 1u;
      const float x = asi + an[w * 32 + b];
      z += __builtin_amdgcn_exp2f(fmaxf(x, ALPHA * x) - m);
    }
  }
  for (int off = 16; off >= 1; off >>= 1) z += __shfl_xor(z, off, 32);
  if (lane == 0) { mrow[h * N_NODES + i] = m; invZ[h * N_NODES + i] = 1.0f / z; }
}

// ------------------------------------------------------------------
// Kernel 4: P.V with WMMA. One wave per (head, 16-row block, column 1/4).
//   - mask words fetched 4 chunks at a time (b128), uniform group skip
//   - B tiles (feats^T bf16) loaded BEFORE the P build to hide L2 latency
//   - P build is fully branchless: p = exp2(lr - m) selected by hoisted bitmask
// ------------------------------------------------------------------
__global__ void __launch_bounds__(32) gat_pv(
    const unsigned* __restrict__ mask, const float* __restrict__ a_s,
    const float* __restrict__ a_n, const float* __restrict__ mrow,
    const float* __restrict__ invZ, const __bf16* __restrict__ featsT,
    float* __restrict__ accbuf) {
  __shared__ float s_an[COLS_PER_PART];
  const int lane   = threadIdx.x;
  const int rowblk = blockIdx.x;           // 0..511
  const int part   = blockIdx.y;           // 0..PARTS-1
  const int h      = blockIdx.z;           // 0..1
  const int i0     = rowblk * 16;
  const int r      = lane & 15;
  const int hi     = lane >> 4;            // 0 or 1 (half-wave)
  const int jbase  = part * COLS_PER_PART;

  for (int k = lane; k < COLS_PER_PART; k += 32)
    s_an[k] = a_n[(size_t)h * N_NODES + jbase + k];
  __syncthreads();

  const float asr = a_s[(size_t)h * N_NODES + i0 + r];
  const float mr  = mrow[(size_t)h * N_NODES + i0 + r];
  float iz[8];
#pragma unroll
  for (int v = 0; v < 8; ++v) iz[v] = invZ[(size_t)h * N_NODES + i0 + v + hi * 8];

  // per-lane bit masks for this lane's 16 K-slots (A-operand layout)
  unsigned bitm[16];
#pragma unroll
  for (int e = 0; e < 16; ++e) {
    const int K = e + (hi << 4);
    bitm[e] = 1u << K;
  }

  v8f acc0 = {}, acc1 = {}, acc2 = {}, acc3 = {};
  const unsigned* mrp = mask + (size_t)(i0 + r) * (N_NODES / 32) + (jbase >> 5);
  const __bf16* fT = featsT + (size_t)h * F_OUT * N_NODES;

  for (int cg = 0; cg < CHUNKS / 4; ++cg) {
    const uint4 mw4 = *reinterpret_cast<const uint4*>(mrp + cg * 4);
    if (__builtin_amdgcn_ballot_w32((mw4.x | mw4.y | mw4.z | mw4.w) != 0u) == 0u)
      continue;                                            // uniform 4-chunk skip
#pragma unroll
    for (int q = 0; q < 4; ++q) {
      const unsigned mw = (q == 0) ? mw4.x : (q == 1) ? mw4.y : (q == 2) ? mw4.z : mw4.w;
      if (__builtin_amdgcn_ballot_w32(mw != 0u) == 0u) continue;  // uniform skip
      const int ch = cg * 4 + q;

      // B operands first: 4 tiles covering F_OUT=64 (32B-aligned contiguous K)
      const size_t bofs = (size_t)(jbase + ch * 32) + (size_t)hi * 16;
      const v16bf b0 = *reinterpret_cast<const v16bf*>(fT + (size_t)(0 * 16 + r) * N_NODES + bofs);
      const v16bf b1 = *reinterpret_cast<const v16bf*>(fT + (size_t)(1 * 16 + r) * N_NODES + bofs);
      const v16bf b2 = *reinterpret_cast<const v16bf*>(fT + (size_t)(2 * 16 + r) * N_NODES + bofs);
      const v16bf b3 = *reinterpret_cast<const v16bf*>(fT + (size_t)(3 * 16 + r) * N_NODES + bofs);

      // a_n values for this lane's K-set: one contiguous 16-float run (b128 DS loads)
      const float4* ap = reinterpret_cast<const float4*>(s_an + ch * 32 + hi * 16);
      const float4 a0 = ap[0], a1 = ap[1], a2 = ap[2], a3 = ap[3];
      const float av[16] = {a0.x, a0.y, a0.z, a0.w, a1.x, a1.y, a1.z, a1.w,
                            a2.x, a2.y, a2.z, a2.w, a3.x, a3.y, a3.z, a3.w};

      // branchless P tile (bf16, A-operand layout)
      v16bf pa;
#pragma unroll
      for (int e = 0; e < 16; ++e) {
        const float x  = asr + av[e];
        const float lr = fmaxf(x, ALPHA * x);
        const float p  = __builtin_amdgcn_exp2f(lr - mr);
        pa[e] = (__bf16)((mw & bitm[e]) ? p : 0.0f);
      }

      acc0 = __builtin_amdgcn_wmma_f32_16x16x32_bf16(false, pa, false, b0, (short)0, acc0, false, false);
      acc1 = __builtin_amdgcn_wmma_f32_16x16x32_bf16(false, pa, false, b1, (short)0, acc1, false, false);
      acc2 = __builtin_amdgcn_wmma_f32_16x16x32_bf16(false, pa, false, b2, (short)0, acc2, false, false);
      acc3 = __builtin_amdgcn_wmma_f32_16x16x32_bf16(false, pa, false, b3, (short)0, acc3, false, false);
    }
  }
  // C/D layout: VGPR v, lanes 0-15 -> row v, lanes 16-31 -> row v+8; col = r + tile*16
#pragma unroll
  for (int v = 0; v < 8; ++v) {
    const int row = i0 + v + hi * 8;
    float* dst = accbuf + (size_t)row * F_OUT + r;
    atomicAdd(dst + 0 * 16, acc0[v] * iz[v]);
    atomicAdd(dst + 1 * 16, acc1[v] * iz[v]);
    atomicAdd(dst + 2 * 16, acc2[v] * iz[v]);
    atomicAdd(dst + 3 * 16, acc3[v] * iz[v]);
  }
}

// ------------------------------------------------------------------
// Kernel 5: out = relu(0.5 * (sum_h nf_h + sum_h bias_h))
// ------------------------------------------------------------------
__global__ void __launch_bounds__(256) gat_final(
    const float* __restrict__ accbuf, const float* __restrict__ biases,
    float* __restrict__ out) {
  const int g = blockIdx.x * 256 + threadIdx.x;
  const int o = g & (F_OUT - 1);
  const float b = biases[o] + biases[F_OUT + o];
  const float v = 0.5f * (accbuf[g] + b);
  out[g] = v > 0.f ? v : 0.f;
}

extern "C" void kernel_launch(void* const* d_in, const int* in_sizes, int n_in,
                              void* d_out, int out_size, void* d_ws, size_t ws_size,
                              hipStream_t stream) {
  const float* X       = (const float*)d_in[0];
  const float* A       = (const float*)d_in[1];
  const float* W       = (const float*)d_in[2];
  const float* biases  = (const float*)d_in[3];
  const float* a_self  = (const float*)d_in[4];
  const float* a_neigh = (const float*)d_in[5];
  float* out = (float*)d_out;

  char* ws = (char*)d_ws;
  __bf16*   featsT = (__bf16*)(ws + OFF_FEATST);
  float*    a_s    = (float*)(ws + OFF_AS);
  float*    a_n    = (float*)(ws + OFF_AN);
  float*    mrow   = (float*)(ws + OFF_M);
  float*    invZ   = (float*)(ws + OFF_IZ);
  unsigned* mask   = (unsigned*)(ws + OFF_MASK);
  float*    accbuf = (float*)(ws + OFF_ACC);

  hipMemsetAsync(accbuf, 0, SZ_ACC, stream);
  gat_proj<<<N_NODES, 128, 0, stream>>>(X, W, a_self, a_neigh, featsT, a_s, a_n);
  gat_mask<<<(N_NODES * (N_NODES / 256)), 256, 0, stream>>>(A, mask);
  gat_rowstats<<<dim3(N_NODES, HEADS), 32, 0, stream>>>(mask, a_s, a_n, mrow, invZ);
  gat_pv<<<dim3(N_NODES / 16, PARTS, HEADS), 32, 0, stream>>>(mask, a_s, a_n, mrow, invZ, featsT, accbuf);
  gat_final<<<(N_NODES * F_OUT) / 256, 256, 0, stream>>>(accbuf, biases, out);
}
